// GNNGuardModel_37460704756551
// MI455X (gfx1250) — matrix operand order repaired and checked by reading, not compiled, not run on hardware
//
#include <hip/hip_runtime.h>
#include <math.h>

#define N_NODES 50000
#define N_EDGES 1600000

typedef __attribute__((ext_vector_type(2)))  float    v2f;
typedef __attribute__((ext_vector_type(8)))  float    v8f;
typedef __attribute__((ext_vector_type(16))) _Float16 v16h;
typedef __attribute__((__vector_size__(16))) int      v4i;

#define AS1 __attribute__((address_space(1)))
#define AS3 __attribute__((address_space(3)))

#if __has_builtin(__builtin_amdgcn_wmma_f32_16x16x4_f32)
#define USE_WMMA_F32 1
#else
#define USE_WMMA_F32 0
#endif

#if __has_builtin(__builtin_amdgcn_global_load_async_to_lds_b128)
#define HAVE_ASYNC_LDS 1
#else
#define HAVE_ASYNC_LDS 0
#endif

__device__ __forceinline__ void gg_wait_asynccnt0() {
#if __has_builtin(__builtin_amdgcn_s_wait_asynccnt)
  __builtin_amdgcn_s_wait_asynccnt(0);
#else
  asm volatile("s_wait_asynccnt 0" ::: "memory");
#endif
}

// ---------------------------------------------------------------- utilities
__global__ __launch_bounds__(256) void gg_zero_kernel(float* __restrict__ p, long n) {
  long i = (long)blockIdx.x * blockDim.x + threadIdx.x;
  if (i < n) p[i] = 0.0f;
}

// -------------------------------------------------- row L2 normalize (wave/row)
__global__ __launch_bounds__(256) void gg_row_normalize(
    const float* __restrict__ x, float* __restrict__ xn, int N, int F) {
  int row  = blockIdx.x * (blockDim.x >> 5) + (threadIdx.x >> 5);
  int lane = threadIdx.x & 31;
  if (row >= N) return;
  const float* a = x + (size_t)row * F;
  float acc = 0.0f;
  for (int f = lane; f < F; f += 32) { float v = a[f]; acc += v * v; }
  for (int off = 16; off; off >>= 1) acc += __shfl_xor(acc, off, 32);
  float inv = (acc > 0.0f) ? (1.0f / sqrtf(acc)) : 0.0f;
  float* o = xn + (size_t)row * F;
  for (int f = lane; f < F; f += 32) o[f] = a[f] * inv;
}

// --------------------- per-edge cosine sim + threshold + rowsum/nz (wave/edge)
__global__ __launch_bounds__(256) void gg_edge_sim(
    const float* __restrict__ xn, const int* __restrict__ src,
    const int* __restrict__ dst, float* __restrict__ w_raw,
    float* __restrict__ rowsum, float* __restrict__ nz, int E, int F) {
  int e    = blockIdx.x * (blockDim.x >> 5) + (threadIdx.x >> 5);
  int lane = threadIdx.x & 31;
  if (e >= E) return;
  int s = src[e], d = dst[e];
  const float* a = xn + (size_t)s * F;
  const float* b = xn + (size_t)d * F;
  float acc = 0.0f;
  for (int f = lane; f < F; f += 32) acc += a[f] * b[f];
  for (int off = 16; off; off >>= 1) acc += __shfl_xor(acc, off, 32);
  if (lane == 0) {
    float w = (acc < 0.5f || s == d) ? 0.0f : acc;   // threshold + drop diagonal
    w_raw[e] = w;
    if (w != 0.0f) {
      atomicAdd(&rowsum[s], w);   // w >= 0 here so |w| == w
      atomicAdd(&nz[s], 1.0f);    // nonzero count (same set after normalize)
    }
  }
}

// --------------------- L1 row-normalize w, accumulate dst degree (thread/edge)
__global__ __launch_bounds__(256) void gg_edge_norm(
    const int* __restrict__ src, const int* __restrict__ dst,
    const float* __restrict__ w_raw, const float* __restrict__ rowsum,
    float* __restrict__ w, float* __restrict__ wdsum, int E) {
  int e = blockIdx.x * blockDim.x + threadIdx.x;
  if (e >= E) return;
  float wr = w_raw[e];
  float rs = rowsum[src[e]];
  float wn = wr / (rs == 0.0f ? 1.0f : rs);
  w[e] = wn;
  if (wn != 0.0f) atomicAdd(&wdsum[dst[e]], wn);
}

// ----------------------------- per-node self_w = 1/(deg+1), dinv = rsqrt(deg2)
__global__ __launch_bounds__(256) void gg_node_deg(
    const float* __restrict__ nz, const float* __restrict__ wdsum,
    float* __restrict__ self_w, float* __restrict__ dinv, int N) {
  int i = blockIdx.x * blockDim.x + threadIdx.x;
  if (i >= N) return;
  float sw   = 1.0f / (nz[i] + 1.0f);
  float dtot = wdsum[i] + sw;
  self_w[i] = sw;
  dinv[i]   = (dtot > 0.0f) ? (1.0f / sqrtf(dtot)) : 0.0f;
}

// ------------------------------------------- H = X @ W  (WMMA, wave per tile)
// One block = 4 waves; all share one column tile tn, waves take 4 consecutive
// row tiles.  B panel (K x 16) is staged in LDS (async-to-LDS when the tile is
// full, zero-padded manual stores otherwise), so the unrolled K-loop is just
// global_load_b64 (A) + ds_load (B) + v_wmma_f32_16x16x4_f32.
// A 16x4: lane%16 = row M, VGPR j holds K = kbase + 2*(lane/16) + j.
// B 4x16:  lane%16 = col N, same K mapping.
// C/D 16x16 f32: VGPR g -> row M = g + 8*(lane/16), col = lane%16.
template <int K>
__global__ __launch_bounds__(128) void gg_gemm_wmma(
    const float* __restrict__ X, const float* __restrict__ W,
    float* __restrict__ H, int Nout, int tiles_n, int tiles_m) {
  __shared__ float ldsB[K * 16];
  const int tid = threadIdx.x;
  const int tn  = blockIdx.x % tiles_n;
  const int tm0 = (blockIdx.x / tiles_n) * 4;

  // ---- stage B panel: columns [tn*16, tn*16+16) of W, K rows
  const bool full_tile = (tn * 16 + 16) <= Nout;
  if (full_tile) {
#if HAVE_ASYNC_LDS
    // 64 contiguous bytes per k-row in both global and LDS -> async b128 copies
    for (int ofs = tid * 16; ofs < K * 64; ofs += 128 * 16) {
      int k  = ofs >> 6;         // row
      int cb = ofs & 63;         // byte within row
      const char* g = (const char*)(W + (size_t)k * Nout + tn * 16) + cb;
      __builtin_amdgcn_global_load_async_to_lds_b128(
          (AS1 v4i*)g, (AS3 v4i*)((char*)ldsB + ofs), 0, 0);
    }
    gg_wait_asynccnt0();
#else
    for (int idx = tid; idx < K * 16; idx += 128) {
      int k = idx >> 4, nn = idx & 15;
      ldsB[idx] = W[(size_t)k * Nout + tn * 16 + nn];
    }
#endif
  } else {
    for (int idx = tid; idx < K * 16; idx += 128) {
      int k = idx >> 4, nn = idx & 15;
      int n = tn * 16 + nn;
      ldsB[idx] = (n < Nout) ? W[(size_t)k * Nout + n] : 0.0f;
    }
  }
  __syncthreads();

  const int wave = tid >> 5;
  const int tm   = tm0 + wave;
  if (tm >= tiles_m) return;                // wave-uniform; EXEC all-1 below
  const int lane = tid & 31;
  const int hi = lane >> 4, lo = lane & 15;
  const float* xrow = X + (size_t)(tm * 16 + lo) * K;

  v8f c = {0.f, 0.f, 0.f, 0.f, 0.f, 0.f, 0.f, 0.f};
#if USE_WMMA_F32
#pragma unroll
  for (int k = 0; k < K; k += 4) {
    int k0 = k + 2 * hi;
    v2f a = *(const v2f*)(xrow + k0);       // 8B aligned: k0 even, row 64B-mult
    v2f b;
    b[0] = ldsB[(k0 + 0) * 16 + lo];
    b[1] = ldsB[(k0 + 1) * 16 + lo];
    c = __builtin_amdgcn_wmma_f32_16x16x4_f32(false, a, false, b,
                                              (short)0, c, false, false);
  }
#else
#pragma unroll
  for (int k = 0; k < K; k += 32) {
    v16h a, b;
#pragma unroll
    for (int j = 0; j < 16; ++j) {
      int ka = k + 8 * hi + j + ((j >= 8) ? 8 : 0);   // A 16x32 f16 layout
      a[j] = (ka < K) ? (_Float16)xrow[ka] : (_Float16)0.0f;
      int kb = k + 8 * hi + j + ((j >= 8) ? 8 : 0);   // B 32x16 f16 layout
      b[j] = (kb < K) ? (_Float16)ldsB[kb * 16 + lo] : (_Float16)0.0f;
    }
    c = __builtin_amdgcn_wmma_f32_16x16x32_f16(false, a, false, b,
                                               (short)0, c, false, false);
  }
#endif
  const int n = tn * 16 + lo;
  if (n < Nout) {
#pragma unroll
    for (int g = 0; g < 8; ++g)
      H[(size_t)(tm * 16 + g + 8 * hi) * Nout + n] = c[g];
  }
}

// ---------------------- scatter-add  agg[dst] += dinv[s]*w*dinv[d] * h[src]
__global__ __launch_bounds__(256) void gg_edge_aggregate(
    const int* __restrict__ src, const int* __restrict__ dst,
    const float* __restrict__ w, const float* __restrict__ dinv,
    const float* __restrict__ h, float* __restrict__ agg, int E, int F) {
  int e    = blockIdx.x * (blockDim.x >> 5) + (threadIdx.x >> 5);
  int lane = threadIdx.x & 31;
  if (e >= E) return;
  float wn = w[e];
  if (wn == 0.0f) return;                 // wave-uniform
  int s = src[e], d = dst[e];
  float coef = dinv[s] * wn * dinv[d];
  const float* hs = h + (size_t)s * F;
  float* od = agg + (size_t)d * F;
  for (int f = lane; f < F; f += 32) atomicAdd(&od[f], coef * hs[f]);
}

// ------------------- out = agg + self_w*dinv^2*h + b ; ReLU (layers 0/1)
__global__ __launch_bounds__(256) void gg_finalize_relu(
    const float* __restrict__ agg, const float* __restrict__ h,
    const float* __restrict__ self_w, const float* __restrict__ dinv,
    const float* __restrict__ bias, float* __restrict__ out, int N, int F) {
  long idx = (long)blockIdx.x * blockDim.x + threadIdx.x;
  if (idx >= (long)N * F) return;
  int i = (int)(idx / F), f = (int)(idx % F);
  float di = dinv[i];
  float v = agg[idx] + self_w[i] * di * di * h[idx] + bias[f];
  out[idx] = v > 0.0f ? v : 0.0f;
}

// ------------------- final layer: + bias + self term, then log_softmax (wave/row)
__global__ __launch_bounds__(256) void gg_finalize_logsoftmax(
    const float* __restrict__ agg, const float* __restrict__ h,
    const float* __restrict__ self_w, const float* __restrict__ dinv,
    const float* __restrict__ bias, float* __restrict__ out, int N, int F) {
  int row  = blockIdx.x * (blockDim.x >> 5) + (threadIdx.x >> 5);
  int lane = threadIdx.x & 31;
  if (row >= N) return;
  float di = dinv[row];
  float sc = self_w[row] * di * di;
  float v[2];
  float mx = -INFINITY;
#pragma unroll
  for (int j = 0; j < 2; ++j) {
    int f = lane + 32 * j;
    float val = -INFINITY;
    if (f < F) val = agg[(size_t)row * F + f] + sc * h[(size_t)row * F + f] + bias[f];
    v[j] = val;
    mx = fmaxf(mx, val);
  }
  for (int off = 16; off; off >>= 1) mx = fmaxf(mx, __shfl_xor(mx, off, 32));
  float se = 0.0f;
#pragma unroll
  for (int j = 0; j < 2; ++j) {
    int f = lane + 32 * j;
    if (f < F) se += expf(v[j] - mx);
  }
  for (int off = 16; off; off >>= 1) se += __shfl_xor(se, off, 32);
  float lse = logf(se);
#pragma unroll
  for (int j = 0; j < 2; ++j) {
    int f = lane + 32 * j;
    if (f < F) out[(size_t)row * F + f] = v[j] - mx - lse;
  }
}

// ---------------------------------------------------------------- host driver
static inline int cdiv(long a, long b) { return (int)((a + b - 1) / b); }

extern "C" void kernel_launch(void* const* d_in, const int* in_sizes, int n_in,
                              void* d_out, int out_size, void* d_ws, size_t ws_size,
                              hipStream_t stream) {
  const float* x0  = (const float*)d_in[0];
  const int*  eix  = (const int*)d_in[1];
  const int*  src  = eix;
  const int*  dst  = eix + N_EDGES;
  const float* Wm[3] = {(const float*)d_in[2], (const float*)d_in[4], (const float*)d_in[6]};
  const float* bm[3] = {(const float*)d_in[3], (const float*)d_in[5], (const float*)d_in[7]};
  float* out = (float*)d_out;

  // workspace layout (floats): ~91 MB total
  float* ws = (float*)d_ws;
  size_t o = 0;
  float* B      = ws + o; o += (size_t)N_NODES * 128;  // xn, then h (reused)
  float* C1     = ws + o; o += (size_t)N_NODES * 128;  // layer1 out / layer3 agg
  float* C2     = ws + o; o += (size_t)N_NODES * 128;  // layer2 out
  float* wraw   = ws + o; o += (size_t)N_EDGES;
  float* wn     = ws + o; o += (size_t)N_EDGES;
  float* rowsum = ws + o; o += (size_t)N_NODES;        // rowsum, nz, wdsum contiguous
  float* nz     = ws + o; o += (size_t)N_NODES;
  float* wdsum  = ws + o; o += (size_t)N_NODES;
  float* selfw  = ws + o; o += (size_t)N_NODES;
  float* dinv   = ws + o; o += (size_t)N_NODES;

  const int WPB = 8;  // waves per 256-thread block
  const int ewaves_blocks = cdiv(N_EDGES, WPB);
  const int nwaves_blocks = cdiv(N_NODES, WPB);
  const int tiles_m = N_NODES / 16;                    // 3125 (exact)

  struct Layer { const float* x; int Fin; int Fout; float* agg; int last; };
  Layer L[3] = {
      {x0, 128, 128, C1, 0},
      {C1, 128, 16,  C2, 0},
      {C2, 16,  40,  C1, 1},
  };

  for (int li = 0; li < 3; ++li) {
    const float* x = L[li].x;
    int Fin = L[li].Fin, Fout = L[li].Fout;
    float* agg = L[li].agg;

    // 1) zero node accumulators (rowsum, nz, wdsum are contiguous)
    gg_zero_kernel<<<cdiv(3L * N_NODES, 256), 256, 0, stream>>>(rowsum, 3L * N_NODES);
    // 2) xn = row-normalize(x)
    gg_row_normalize<<<nwaves_blocks, 256, 0, stream>>>(x, B, N_NODES, Fin);
    // 3) per-edge sim -> w_raw, rowsum, nz
    gg_edge_sim<<<ewaves_blocks, 256, 0, stream>>>(B, src, dst, wraw, rowsum, nz,
                                                   N_EDGES, Fin);
    // 4) L1 normalize, accumulate dst-degree
    gg_edge_norm<<<cdiv(N_EDGES, 256), 256, 0, stream>>>(src, dst, wraw, rowsum,
                                                         wn, wdsum, N_EDGES);
    // 5) self_w, dinv
    gg_node_deg<<<cdiv(N_NODES, 256), 256, 0, stream>>>(nz, wdsum, selfw, dinv,
                                                        N_NODES);
    // 6) h = x @ W   (WMMA; overwrites xn buffer, which is no longer needed)
    int tiles_n = cdiv(Fout, 16);
    int gemm_blocks = cdiv(tiles_m, 4) * tiles_n;
    if (Fin == 128) {
      gg_gemm_wmma<128><<<gemm_blocks, 128, 0, stream>>>(x, Wm[li], B, Fout,
                                                         tiles_n, tiles_m);
    } else {
      gg_gemm_wmma<16><<<gemm_blocks, 128, 0, stream>>>(x, Wm[li], B, Fout,
                                                        tiles_n, tiles_m);
    }
    // 7) zero aggregation buffer
    gg_zero_kernel<<<cdiv((long)N_NODES * Fout, 256), 256, 0, stream>>>(
        agg, (long)N_NODES * Fout);
    // 8) agg[dst] += dinv[s]*w*dinv[d] * h[src]
    gg_edge_aggregate<<<ewaves_blocks, 256, 0, stream>>>(src, dst, wn, dinv, B,
                                                         agg, N_EDGES, Fout);
    // 9) finalize
    if (!L[li].last) {
      gg_finalize_relu<<<cdiv((long)N_NODES * Fout, 256), 256, 0, stream>>>(
          agg, B, selfw, dinv, bm[li], agg, N_NODES, Fout);
    } else {
      gg_finalize_logsoftmax<<<nwaves_blocks, 256, 0, stream>>>(
          agg, B, selfw, dinv, bm[li], out, N_NODES, Fout);
    }
  }
}
